// Encoder_60473139528257
// MI455X (gfx1250) — matrix operand order repaired and checked by reading, not compiled, not run on hardware
//
#include <hip/hip_runtime.h>
#include <hip/hip_bf16.h>

// ---------------------------------------------------------------------------
// CDNA5 (gfx1250) wave32 WMMA implementation of the 6-layer relative-position
// transformer encoder.  All GEMM-shaped work -> v_wmma_f32_16x16x32_f16.
// Relative-position einsums reduced to a 9-tap band (W=4) around the diagonal.
// ---------------------------------------------------------------------------

typedef __attribute__((ext_vector_type(16))) _Float16 v16h;
typedef __attribute__((ext_vector_type(8)))  float    v8f;

constexpr int BATCH = 8;
constexpr int CH    = 192;   // channels
constexpr int TT    = 1024;  // time steps
constexpr int FF    = 768;   // filter channels
constexpr int NH    = 2;     // heads
constexpr int NL    = 6;     // layers
constexpr int WIN   = 4;     // rel-attn window
constexpr int KF    = 3;     // conv kernel size
constexpr int DH    = 96;    // head dim
constexpr float QSCALE = 0.10206207261596575f; // 1/sqrt(96)

// ---------------- WMMA helpers (wave32, 16x16x32 f16 -> f32) ----------------

__device__ __forceinline__ v8f wmma_f16(v16h a, v16h b, v8f c) {
  // (neg_a, A, neg_b, B, c_mod, C, reuse_a, reuse_b)
  return __builtin_amdgcn_wmma_f32_16x16x32_f16(false, a, false, b,
                                                (short)0, c, false, false);
}

// A fragment: 16x32 f16, LDS tile row-major [16][lda].
// lanes 0-15 row M=lane, K=0..15;
// lanes 16-31 row M=lane-16, K=16..31.
__device__ __forceinline__ v16h load_afrag(const _Float16* A, int lda,
                                           int r0, int lane) {
  const int row = r0 + (lane & 15);
  const int ko  = (lane >> 4) * 16;
  const _Float16* p = A + row * lda + ko;
  v16h a;
#pragma unroll
  for (int j = 0; j < 16; ++j) a[j] = p[j];
  return a;
}

// B fragment: 32x16 f16 with the tile stored TRANSPOSED in LDS as Bt[N][K].
// lanes 0-15 col N=lane, K=0..15; lanes 16-31 col N=lane-16, K=16..31.
__device__ __forceinline__ v16h load_bfrag(const _Float16* Bt, int ldb,
                                           int n0, int lane) {
  const int n  = n0 + (lane & 15);
  const int ko = (lane >> 4) * 16;
  const _Float16* p = Bt + n * ldb + ko;
  v16h b;
#pragma unroll
  for (int j = 0; j < 16; ++j) b[j] = p[j];
  return b;
}

// ------------------------- QKV projection (fused z) -------------------------
// grid (128, 3, 3): 64-col x 64-row tiles over [C=192, N=B*T=8192], z = q/k/v.
// out: f16 [b][h][t][d]; q pre-scaled by 1/sqrt(D).
__global__ __launch_bounds__(256) void k_qkv(
    const float* __restrict__ x,
    const float* __restrict__ Wq, const float* __restrict__ bq,
    const float* __restrict__ Wk, const float* __restrict__ bk,
    const float* __restrict__ Wv, const float* __restrict__ bv,
    _Float16* __restrict__ qo, _Float16* __restrict__ ko_,
    _Float16* __restrict__ vo) {
  __shared__ _Float16 Ash[64 * 40];
  __shared__ _Float16 Bsh[64 * 40];
  const int z = blockIdx.z;
  const float* Wm = (z == 0) ? Wq : (z == 1) ? Wk : Wv;
  const float* bb = (z == 0) ? bq : (z == 1) ? bk : bv;
  _Float16* outp  = (z == 0) ? qo : (z == 1) ? ko_ : vo;
  const int n0 = blockIdx.x * 64, m0 = blockIdx.y * 64;
  const int tid = threadIdx.x, lane = tid & 31, wave = tid >> 5;
  const int wm = wave & 3, wn = wave >> 2;   // 4x2 wave tile grid
  v8f c0 = {}, c1 = {};
  for (int k0 = 0; k0 < CH; k0 += 32) {
    { // stage A = weights (row-major [out_c][in_c])
      const int row = tid >> 2, kc = (tid & 3) * 8;
      const float* src = &Wm[(size_t)(m0 + row) * CH + k0 + kc];
      if (k0 + 32 < CH) __builtin_prefetch(src + 32, 0, 1);
      _Float16* dst = &Ash[row * 40 + kc];
#pragma unroll
      for (int j = 0; j < 8; ++j) dst[j] = (_Float16)src[j];
    }
    { // stage Bt[n][k] = x[b][c=k][t]
      const int ln = tid >> 2, kc = (tid & 3) * 8;
      const int n = n0 + ln, b = n >> 10, t = n & 1023;
      const float* src = &x[((size_t)b * CH + k0 + kc) * TT + t];
      _Float16* dst = &Bsh[ln * 40 + kc];
#pragma unroll
      for (int j = 0; j < 8; ++j) dst[j] = (_Float16)src[(size_t)j * TT];
    }
    __syncthreads();
    const v16h a  = load_afrag(Ash, 40, wm * 16, lane);
    const v16h b0 = load_bfrag(Bsh, 40, wn * 32, lane);
    const v16h b1 = load_bfrag(Bsh, 40, wn * 32 + 16, lane);
    c0 = wmma_f16(a, b0, c0);
    c1 = wmma_f16(a, b1, c1);
    __syncthreads();
  }
  const float scl = (z == 0) ? QSCALE : 1.0f;
  const int mb = m0 + wm * 16 + ((lane >> 4) << 3);
  const int nb = n0 + wn * 32 + (lane & 15);
#pragma unroll
  for (int r = 0; r < 8; ++r) {
    const int m = mb + r;
    const float bias = bb[m];
    const int h = m / DH, d = m - h * DH;
#pragma unroll
    for (int sub = 0; sub < 2; ++sub) {
      const int n = nb + sub * 16, b = n >> 10, t = n & 1023;
      const float vv = ((sub ? c1[r] : c0[r]) + bias) * scl;
      outp[(((size_t)b * NH + h) * TT + t) * DH + d] = (_Float16)vv;
    }
  }
}

// ------------- rel-K band: band[bh][t][r] = qs[t]·rel_k[r], r=0..8 ----------
__global__ __launch_bounds__(256) void k_band(const _Float16* __restrict__ q,
                                              const float* __restrict__ rk,
                                              float* __restrict__ band) {
  const int i = blockIdx.x * 256 + threadIdx.x;   // bh*T*9
  const int r = i % 9;
  const int t = (i / 9) % TT;
  const int bh = i / (9 * TT);
  const _Float16* qp = q + ((size_t)bh * TT + t) * DH;
  const float* rp = rk + r * DH;
  float s = 0.f;
  for (int d = 0; d < DH; ++d) s += (float)qp[d] * rp[d];
  band[i] = s;
}

// --------- scores: P[bh][t][s] = qs·kᵀ + band(s-t) with mask, f16 out -------
// grid (16 s-tiles, 16 t-tiles, 16 bh)
__global__ __launch_bounds__(256) void k_scores(
    const _Float16* __restrict__ q, const _Float16* __restrict__ kk,
    const float* __restrict__ band, const float* __restrict__ mask,
    _Float16* __restrict__ P) {
  __shared__ _Float16 Ash[64 * 40];
  __shared__ _Float16 Bsh[64 * 40];
  const int bh = blockIdx.z, b = bh >> 1;
  const int s0 = blockIdx.x * 64, t0 = blockIdx.y * 64;
  const int tid = threadIdx.x, lane = tid & 31, wave = tid >> 5;
  const int wm = wave & 3, wn = wave >> 2;
  v8f c0 = {}, c1 = {};
  for (int k0 = 0; k0 < DH; k0 += 32) {
    {
      const int row = tid >> 2, kc = (tid & 3) * 8;
      const _Float16* src = q + ((size_t)bh * TT + t0 + row) * DH + k0 + kc;
      _Float16* dst = &Ash[row * 40 + kc];
#pragma unroll
      for (int j = 0; j < 8; ++j) dst[j] = src[j];
    }
    {
      const int row = tid >> 2, kc = (tid & 3) * 8;
      const _Float16* src = kk + ((size_t)bh * TT + s0 + row) * DH + k0 + kc;
      _Float16* dst = &Bsh[row * 40 + kc];
#pragma unroll
      for (int j = 0; j < 8; ++j) dst[j] = src[j];
    }
    __syncthreads();
    const v16h a  = load_afrag(Ash, 40, wm * 16, lane);
    const v16h b0 = load_bfrag(Bsh, 40, wn * 32, lane);
    const v16h b1 = load_bfrag(Bsh, 40, wn * 32 + 16, lane);
    c0 = wmma_f16(a, b0, c0);
    c1 = wmma_f16(a, b1, c1);
    __syncthreads();
  }
  const int mb = t0 + wm * 16 + ((lane >> 4) << 3);
  const int nb = s0 + wn * 32 + (lane & 15);
#pragma unroll
  for (int r = 0; r < 8; ++r) {
    const int i = mb + r;
    const float mi = mask[b * TT + i];
#pragma unroll
    for (int sub = 0; sub < 2; ++sub) {
      const int s = nb + sub * 16;
      float v = sub ? c1[r] : c0[r];
      const int ri = s - i + WIN;
      if (ri >= 0 && ri <= 2 * WIN) v += band[((size_t)bh * TT + i) * 9 + ri];
      if (mi == 0.f || mask[b * TT + s] == 0.f) v = -10000.f;
      P[((size_t)bh * TT + i) * TT + s] = (_Float16)v;
    }
  }
}

// --------------------------- row softmax (in place) -------------------------
// one 256-thread block per row of 1024, grid = B*H*T
__global__ __launch_bounds__(256) void k_softmax(_Float16* __restrict__ P) {
  _Float16* p = P + (size_t)blockIdx.x * TT;
  const int tid = threadIdx.x;
  __shared__ float red[256];
  float vals[4];
  float mx = -1e30f;
#pragma unroll
  for (int j = 0; j < 4; ++j) {
    vals[j] = (float)p[tid + j * 256];
    mx = fmaxf(mx, vals[j]);
  }
  red[tid] = mx; __syncthreads();
  for (int s = 128; s > 0; s >>= 1) {
    if (tid < s) red[tid] = fmaxf(red[tid], red[tid + s]);
    __syncthreads();
  }
  mx = red[0]; __syncthreads();
  float sum = 0.f;
#pragma unroll
  for (int j = 0; j < 4; ++j) { vals[j] = __expf(vals[j] - mx); sum += vals[j]; }
  red[tid] = sum; __syncthreads();
  for (int s = 128; s > 0; s >>= 1) {
    if (tid < s) red[tid] += red[tid + s];
    __syncthreads();
  }
  const float inv = 1.f / red[0];
#pragma unroll
  for (int j = 0; j < 4; ++j) p[tid + j * 256] = (_Float16)(vals[j] * inv);
}

// ------------------- AV: out[b][h*D+d][t] = P·v  (WMMA) ---------------------
// grid (3 d-tiles of 32, 16 t-tiles of 64, 16 bh)
__global__ __launch_bounds__(256) void k_av(const _Float16* __restrict__ P,
                                            const _Float16* __restrict__ v,
                                            float* __restrict__ attn) {
  __shared__ _Float16 Ash[64 * 40];
  __shared__ _Float16 Bsh[32 * 40];
  const int bh = blockIdx.z, b = bh >> 1, h = bh & 1;
  const int d0 = blockIdx.x * 32, t0 = blockIdx.y * 64;
  const int tid = threadIdx.x, lane = tid & 31, wave = tid >> 5;
  const int wm = wave & 3, wn = wave >> 2;
  v8f c0 = {};
  for (int k0 = 0; k0 < TT; k0 += 32) {
    { // A = probabilities (already f16, contiguous)
      const int row = tid >> 2, kc = (tid & 3) * 8;
      const _Float16* src = P + ((size_t)bh * TT + t0 + row) * TT + k0 + kc;
      if (k0 + 32 < TT) __builtin_prefetch(src + 32, 0, 1);
      _Float16* dst = &Ash[row * 40 + kc];
#pragma unroll
      for (int j = 0; j < 8; ++j) dst[j] = src[j];
    }
    { // Bt[d][s] = v[s][d] (transpose gather)
      const int ln = tid >> 3, kc = (tid & 7) * 4;
      _Float16* dst = &Bsh[ln * 40 + kc];
#pragma unroll
      for (int j = 0; j < 4; ++j)
        dst[j] = v[((size_t)bh * TT + k0 + kc + j) * DH + d0 + ln];
    }
    __syncthreads();
    const v16h a  = load_afrag(Ash, 40, wm * 16, lane);
    const v16h bf = load_bfrag(Bsh, 40, wn * 16, lane);
    c0 = wmma_f16(a, bf, c0);
    __syncthreads();
  }
  const int mb = t0 + wm * 16 + ((lane >> 4) << 3);
  const int d  = d0 + wn * 16 + (lane & 15);
#pragma unroll
  for (int r = 0; r < 8; ++r) {
    const int t = mb + r;
    attn[((size_t)b * CH + h * DH + d) * TT + t] = c0[r];
  }
}

// ------------ AV band term: out += Σ_r P[t, t+r-W] * rel_v[r][d] ------------
__global__ __launch_bounds__(256) void k_avband(const _Float16* __restrict__ P,
                                                const float* __restrict__ rv,
                                                float* __restrict__ attn) {
  const int i = blockIdx.x * 256 + threadIdx.x;   // (bh, t, d)
  const int d = i % DH;
  const int t = (i / DH) % TT;
  const int bh = i / (DH * TT);
  const int b = bh >> 1, h = bh & 1;
  const _Float16* prow = P + ((size_t)bh * TT + t) * TT;
  float s = 0.f;
#pragma unroll
  for (int r = 0; r <= 2 * WIN; ++r) {
    const int si = t + r - WIN;
    if (si >= 0 && si < TT) s += (float)prow[si] * rv[r * DH + d];
  }
  attn[((size_t)b * CH + h * DH + d) * TT + t] += s;
}

// ------------------------- output projection (WMMA) -------------------------
// grid (128, 3): y = Wo·attn + bo, f32 out [B][C][T]
__global__ __launch_bounds__(256) void k_oproj(const float* __restrict__ attn,
                                               const float* __restrict__ Wo,
                                               const float* __restrict__ bo,
                                               float* __restrict__ yout) {
  __shared__ _Float16 Ash[64 * 40];
  __shared__ _Float16 Bsh[64 * 40];
  const int n0 = blockIdx.x * 64, m0 = blockIdx.y * 64;
  const int tid = threadIdx.x, lane = tid & 31, wave = tid >> 5;
  const int wm = wave & 3, wn = wave >> 2;
  v8f c0 = {}, c1 = {};
  for (int k0 = 0; k0 < CH; k0 += 32) {
    {
      const int row = tid >> 2, kc = (tid & 3) * 8;
      const float* src = &Wo[(size_t)(m0 + row) * CH + k0 + kc];
      _Float16* dst = &Ash[row * 40 + kc];
#pragma unroll
      for (int j = 0; j < 8; ++j) dst[j] = (_Float16)src[j];
    }
    {
      const int ln = tid >> 2, kc = (tid & 3) * 8;
      const int n = n0 + ln, b = n >> 10, t = n & 1023;
      const float* src = &attn[((size_t)b * CH + k0 + kc) * TT + t];
      _Float16* dst = &Bsh[ln * 40 + kc];
#pragma unroll
      for (int j = 0; j < 8; ++j) dst[j] = (_Float16)src[(size_t)j * TT];
    }
    __syncthreads();
    const v16h a  = load_afrag(Ash, 40, wm * 16, lane);
    const v16h b0 = load_bfrag(Bsh, 40, wn * 32, lane);
    const v16h b1 = load_bfrag(Bsh, 40, wn * 32 + 16, lane);
    c0 = wmma_f16(a, b0, c0);
    c1 = wmma_f16(a, b1, c1);
    __syncthreads();
  }
  const int mb = m0 + wm * 16 + ((lane >> 4) << 3);
  const int nb = n0 + wn * 32 + (lane & 15);
#pragma unroll
  for (int r = 0; r < 8; ++r) {
    const int m = mb + r;
    const float bias = bo[m];
#pragma unroll
    for (int sub = 0; sub < 2; ++sub) {
      const int n = nb + sub * 16, b = n >> 10, t = n & 1023;
      yout[((size_t)b * CH + m) * TT + t] = (sub ? c1[r] : c0[r]) + bias;
    }
  }
}

// ----------------- K=3 conv as tap-accumulated WMMA GEMM --------------------
// kk = tap*CIN + c;  input column shifted by tap-1 with zero padding.
template <int CIN>
__global__ __launch_bounds__(256) void k_conv(
    const float* __restrict__ xin, const float* __restrict__ mask,
    const float* __restrict__ Wf, const float* __restrict__ bias,
    float* __restrict__ yout, int cout, int do_relu, int do_maskout) {
  __shared__ _Float16 Ash[64 * 40];
  __shared__ _Float16 Bsh[64 * 40];
  const int n0 = blockIdx.x * 64, m0 = blockIdx.y * 64;
  const int tid = threadIdx.x, lane = tid & 31, wave = tid >> 5;
  const int wm = wave & 3, wn = wave >> 2;
  v8f c0 = {}, c1 = {};
  for (int k0 = 0; k0 < KF * CIN; k0 += 32) {
    { // A: W[f][c][tap]
      const int row = tid >> 2, kc = (tid & 3) * 8;
      _Float16* dst = &Ash[row * 40 + kc];
#pragma unroll
      for (int j = 0; j < 8; ++j) {
        const int kx = k0 + kc + j, tap = kx / CIN, c = kx - tap * CIN;
        dst[j] = (_Float16)Wf[((size_t)(m0 + row) * CIN + c) * KF + tap];
      }
    }
    { // Bt[n][kk] = masked, tap-shifted input
      const int ln = tid >> 2, kc = (tid & 3) * 8;
      const int n = n0 + ln, b = n >> 10, t = n & 1023;
      _Float16* dst = &Bsh[ln * 40 + kc];
#pragma unroll
      for (int j = 0; j < 8; ++j) {
        const int kx = k0 + kc + j, tap = kx / CIN, c = kx - tap * CIN;
        const int tp = t + tap - 1;
        float vv = 0.f;
        if (tp >= 0 && tp < TT)
          vv = xin[((size_t)b * CIN + c) * TT + tp] * mask[b * TT + tp];
        dst[j] = (_Float16)vv;
      }
    }
    __syncthreads();
    const v16h a  = load_afrag(Ash, 40, wm * 16, lane);
    const v16h b0 = load_bfrag(Bsh, 40, wn * 32, lane);
    const v16h b1 = load_bfrag(Bsh, 40, wn * 32 + 16, lane);
    c0 = wmma_f16(a, b0, c0);
    c1 = wmma_f16(a, b1, c1);
    __syncthreads();
  }
  const int mb = m0 + wm * 16 + ((lane >> 4) << 3);
  const int nb = n0 + wn * 32 + (lane & 15);
#pragma unroll
  for (int r = 0; r < 8; ++r) {
    const int m = mb + r;
    const float bi = bias[m];
#pragma unroll
    for (int sub = 0; sub < 2; ++sub) {
      const int n = nb + sub * 16, b = n >> 10, t = n & 1023;
      float vv = (sub ? c1[r] : c0[r]) + bi;
      if (do_relu) vv = fmaxf(vv, 0.f);
      if (do_maskout) vv *= mask[b * TT + t];
      yout[((size_t)b * cout + m) * TT + t] = vv;
    }
  }
}

// ---------------------- residual add + LayerNorm over C ---------------------
// one block per (b, t) column; in-place safe (xout may alias xin).
__global__ __launch_bounds__(256) void k_addln(const float* __restrict__ xin,
                                               const float* __restrict__ yin,
                                               const float* __restrict__ g,
                                               const float* __restrict__ bta,
                                               float* __restrict__ xout) {
  const int col = blockIdx.x, b = col >> 10, t = col & 1023;
  const int tid = threadIdx.x;
  __shared__ float red[256];
  float myv = 0.f;
  if (tid < CH)
    myv = xin[((size_t)b * CH + tid) * TT + t] +
          yin[((size_t)b * CH + tid) * TT + t];
  red[tid] = (tid < CH) ? myv : 0.f;
  __syncthreads();
  for (int s = 128; s > 0; s >>= 1) {
    if (tid < s) red[tid] += red[tid + s];
    __syncthreads();
  }
  const float mean = red[0] * (1.f / CH);
  __syncthreads();
  const float dv = (tid < CH) ? (myv - mean) : 0.f;
  red[tid] = dv * dv;
  __syncthreads();
  for (int s = 128; s > 0; s >>= 1) {
    if (tid < s) red[tid] += red[tid + s];
    __syncthreads();
  }
  const float rs = rsqrtf(red[0] * (1.f / CH) + 1e-5f);
  if (tid < CH)
    xout[((size_t)b * CH + tid) * TT + t] = dv * rs * g[tid] + bta[tid];
}

// ------------------------------ elementwise ---------------------------------
__global__ __launch_bounds__(256) void k_maskmul(const float* __restrict__ x,
                                                 const float* __restrict__ mask,
                                                 float* __restrict__ o) {
  const int i = blockIdx.x * 256 + threadIdx.x;
  const int t = i & 1023;
  const int b = i / (CH * TT);
  o[i] = x[i] * mask[b * TT + t];
}

// ------------------------------- launcher -----------------------------------

extern "C" void kernel_launch(void* const* d_in, const int* in_sizes, int n_in,
                              void* d_out, int out_size, void* d_ws,
                              size_t ws_size, hipStream_t stream) {
  (void)in_sizes; (void)n_in; (void)out_size; (void)ws_size;

  const float* x    = (const float*)d_in[0];
  const float* mask = (const float*)d_in[1];
  const float* Wq  = (const float*)d_in[2];  const float* bq  = (const float*)d_in[3];
  const float* Wk  = (const float*)d_in[4];  const float* bk  = (const float*)d_in[5];
  const float* Wv  = (const float*)d_in[6];  const float* bv  = (const float*)d_in[7];
  const float* Wo  = (const float*)d_in[8];  const float* bo  = (const float*)d_in[9];
  const float* rk  = (const float*)d_in[10]; const float* rv  = (const float*)d_in[11];
  const float* g1  = (const float*)d_in[12]; const float* b1  = (const float*)d_in[13];
  const float* Wf1 = (const float*)d_in[14]; const float* bf1 = (const float*)d_in[15];
  const float* Wf2 = (const float*)d_in[16]; const float* bf2 = (const float*)d_in[17];
  const float* g2  = (const float*)d_in[18]; const float* b2  = (const float*)d_in[19];

  // workspace partition
  constexpr size_t SZ_X    = (size_t)BATCH * CH * TT * 4;        // 6.29 MB
  constexpr size_t SZ_Q    = (size_t)BATCH * NH * TT * DH * 2;   // 3.15 MB
  constexpr size_t SZ_BAND = (size_t)BATCH * NH * TT * 9 * 4;    // 0.59 MB
  constexpr size_t SZ_P    = (size_t)BATCH * NH * TT * TT * 2;   // 33.5 MB
  constexpr size_t SZ_F    = (size_t)BATCH * FF * TT * 4;        // 25.2 MB
  char* ws = (char*)d_ws;
  float*    bx   = (float*)(ws);
  _Float16* qb   = (_Float16*)(ws + SZ_X);
  _Float16* kb   = (_Float16*)(ws + SZ_X + SZ_Q);
  _Float16* vb   = (_Float16*)(ws + SZ_X + 2 * SZ_Q);
  float*    band = (float*)(ws + SZ_X + 3 * SZ_Q);
  _Float16* Pb   = (_Float16*)(ws + SZ_X + 3 * SZ_Q + SZ_BAND);
  float*    ab   = (float*)(ws + SZ_X + 3 * SZ_Q + SZ_BAND + SZ_P);
  float*    yb   = (float*)(ws + SZ_X + 3 * SZ_Q + SZ_BAND + SZ_P + SZ_X);
  float*    fb   = (float*)(ws + SZ_X + 3 * SZ_Q + SZ_BAND + SZ_P + 2 * SZ_X);
  (void)SZ_F;

  const int NTOT = BATCH * CH * TT;            // 1,572,864
  k_maskmul<<<NTOT / 256, 256, 0, stream>>>(x, mask, bx);

  for (int i = 0; i < NL; ++i) {
    // --- MHA ---
    k_qkv<<<dim3(128, 3, 3), 256, 0, stream>>>(
        bx, Wq + (size_t)i * CH * CH, bq + (size_t)i * CH,
        Wk + (size_t)i * CH * CH, bk + (size_t)i * CH,
        Wv + (size_t)i * CH * CH, bv + (size_t)i * CH, qb, kb, vb);
    k_band<<<(BATCH * NH * TT * 9) / 256, 256, 0, stream>>>(
        qb, rk + (size_t)i * 9 * DH, band);
    k_scores<<<dim3(16, 16, 16), 256, 0, stream>>>(qb, kb, band, mask, Pb);
    k_softmax<<<BATCH * NH * TT, 256, 0, stream>>>(Pb);
    k_av<<<dim3(3, 16, 16), 256, 0, stream>>>(Pb, vb, ab);
    k_avband<<<(BATCH * NH * TT * DH) / 256, 256, 0, stream>>>(
        Pb, rv + (size_t)i * 9 * DH, ab);
    k_oproj<<<dim3(128, 3), 256, 0, stream>>>(
        ab, Wo + (size_t)i * CH * CH, bo + (size_t)i * CH, yb);
    k_addln<<<BATCH * TT, 256, 0, stream>>>(bx, yb, g1 + (size_t)i * CH,
                                            b1 + (size_t)i * CH, bx);
    // --- FFN (K=3 convs as WMMA GEMMs) ---
    k_conv<CH><<<dim3(128, FF / 64), 256, 0, stream>>>(
        bx, mask, Wf1 + (size_t)i * FF * CH * KF, bf1 + (size_t)i * FF, fb,
        FF, /*relu=*/1, /*maskout=*/0);
    k_conv<FF><<<dim3(128, CH / 64), 256, 0, stream>>>(
        fb, mask, Wf2 + (size_t)i * CH * FF * KF, bf2 + (size_t)i * CH, yb,
        CH, /*relu=*/0, /*maskout=*/1);
    k_addln<<<BATCH * TT, 256, 0, stream>>>(bx, yb, g2 + (size_t)i * CH,
                                            b2 + (size_t)i * CH, bx);
  }

  k_maskmul<<<NTOT / 256, 256, 0, stream>>>(bx, mask, (float*)d_out);
}
